// LA_8289286881311
// MI455X (gfx1250) — matrix operand order repaired and checked by reading, not compiled, not run on hardware
//
#include <hip/hip_runtime.h>
#include <hip/hip_bf16.h>

typedef __attribute__((ext_vector_type(16))) __bf16 v16bf;
typedef __attribute__((ext_vector_type(8)))  __bf16 v8bf;
typedef __attribute__((ext_vector_type(8)))  float  v8f;

union BF16x8 { uint4 u; v8bf v; };

// ---- Tensor Data Mover availability (6-arg amdgpu-toolchain form) ----
#if defined(__has_builtin)
#  if __has_builtin(__builtin_amdgcn_tensor_load_to_lds) && \
      __has_builtin(__builtin_amdgcn_s_wait_tensorcnt)
#    define USE_TDM 1
#  endif
#endif
#ifndef USE_TDM
#  define USE_TDM 0
#endif

#if USE_TDM
typedef unsigned int tdm_v4u __attribute__((ext_vector_type(4)));
typedef int          tdm_v8i __attribute__((ext_vector_type(8)));
typedef int          tdm_v4i __attribute__((ext_vector_type(4)));

// Low 32 bits of a flat shared-memory address are the LDS byte offset
// (ISA 10.2).
__device__ __forceinline__ unsigned lds_off(const void* p) {
    return (unsigned)(unsigned long long)p;
}

// 2-D TDM tile load: tile_d0 elements (bf16) per row, tile_d1 rows,
// row stride stride0 (elements), LDS rows padded per pad_interval/pad_amount.
__device__ __forceinline__ void tdm_load_2d(unsigned lds_addr, const void* gaddr,
                                            unsigned tensor_d0, unsigned tensor_d1,
                                            unsigned stride0,
                                            unsigned tile_d0, unsigned tile_d1,
                                            unsigned pad_interval, unsigned pad_amount) {
    unsigned long long ga = (unsigned long long)gaddr;
    tdm_v4u g0;
    g0.x = 1u;                                           // count=1 (valid user D#)
    g0.y = lds_addr;                                     // LDS byte address
    g0.z = (unsigned)ga;                                 // global_addr[31:0]
    g0.w = (unsigned)((ga >> 32) & 0x1FFFFFFu) | (2u << 30);  // addr[56:32] | type=2
    tdm_v8i g1;
    g1[0] = (int)((1u << 16)                             // data_size=1 -> 2 bytes
                | (1u << 20)                             // pad_enable
                | (pad_interval << 22) | (pad_amount << 25));
    g1[1] = (int)((tensor_d0 & 0xFFFFu) << 16);          // tensor_dim0[15:0] @ bit48
    g1[2] = (int)(((tensor_d0 >> 16) & 0xFFFFu) | ((tensor_d1 & 0xFFFFu) << 16));
    g1[3] = (int)(((tensor_d1 >> 16) & 0xFFFFu) | (tile_d0 << 16));
    g1[4] = (int)(tile_d1 & 0xFFFFu);                    // tile_dim1 (tile_dim2=0)
    g1[5] = (int)stride0;                                // tensor_dim0_stride[31:0]
    g1[6] = 0;
    g1[7] = 0;                                           // dim1_stride unused (2-D)
    tdm_v4i z4 = {0, 0, 0, 0};
    tdm_v8i z8 = {0, 0, 0, 0, 0, 0, 0, 0};
    __builtin_amdgcn_tensor_load_to_lds(g0, g1, z4, z4, z8, 0);
}
#endif

// Build a 16-element WMMA fragment from two 16-byte LDS reads.
__device__ __forceinline__ v16bf make_frag(const __bf16* p0, const __bf16* p1) {
    BF16x8 a, b;
    a.u = *(const uint4*)p0;
    b.u = *(const uint4*)p1;
    v16bf f;
#pragma unroll
    for (int i = 0; i < 8; ++i) { f[i] = a.v[i]; f[i + 8] = b.v[i]; }
    return f;
}

__device__ __forceinline__ float sigmoidf_(float x) {
    return 1.f / (1.f + __expf(-x));
}

#define CDIM 768
#define NTOK 1024
#define BATCH 8
#define HEADS 12
#define HD 64

// ---------------------------------------------------------------- cvt fp32 -> bf16
__global__ __launch_bounds__(256)
void cvt_bf16_kernel(const float* __restrict__ in, __bf16* __restrict__ out, int n) {
    int i = blockIdx.x * 256 + threadIdx.x;
    int stride = gridDim.x * 256;
    for (; i < n; i += stride) out[i] = (__bf16)in[i];
}

// ---------------------------------------------------------------- y transpose + LayerNorm -> y2 (bf16 [B*N, C])
__global__ __launch_bounds__(256)
void ln_kernel(const float* __restrict__ y, const float* __restrict__ g,
               const float* __restrict__ be, __bf16* __restrict__ y2b) {
    __shared__ float red[8][32], red2[8][32];
    __shared__ float s_mu[32], s_rs[32];
    __shared__ __bf16 s_raw[32 * CDIM];
    const int t = threadIdx.x;
    const int b = blockIdx.x >> 5;
    const int n0 = (blockIdx.x & 31) * 32;
    const int myn = t & 31, cs = t >> 5;

    float sum = 0.f, sum2 = 0.f;
    for (int c = cs; c < CDIM; c += 8) {
        float v = y[((long)b * CDIM + c) * NTOK + n0 + myn];
        s_raw[myn * CDIM + c] = (__bf16)v;
        sum += v; sum2 += v * v;
    }
    red[cs][myn] = sum; red2[cs][myn] = sum2;
    __syncthreads();
    if (t < 32) {
        float s = 0.f, s2 = 0.f;
#pragma unroll
        for (int i = 0; i < 8; ++i) { s += red[i][t]; s2 += red2[i][t]; }
        float mu = s * (1.f / CDIM);
        float var = s2 * (1.f / CDIM) - mu * mu;
        s_mu[t] = mu;
        s_rs[t] = rsqrtf(var + 1e-5f);
    }
    __syncthreads();
    for (int idx = t; idx < 32 * CDIM; idx += 256) {
        int n = idx / CDIM, c = idx - n * CDIM;
        float v = (float)s_raw[idx];
        float o = (v - s_mu[n]) * s_rs[n] * g[c] + be[c];
        y2b[((long)b * NTOK + n0 + n) * CDIM + c] = (__bf16)o;
    }
}

// ---------------------------------------------------------------- fused QKV GEMM + sigmoid gating
// O1 = x @ W^T, O2 = y2 @ W^T with W = [q_w; kv_w] bf16, shared B tile; double-buffered LDS,
// next tiles prefetched by the Tensor Data Mover while WMMAs run on the current buffer.
__global__ __launch_bounds__(256)
void gemm_qkv_kernel(const __bf16* __restrict__ xb, const __bf16* __restrict__ y2b,
                     const __bf16* __restrict__ wb,
                     const float* __restrict__ qbias, const float* __restrict__ kvbias,
                     __bf16* __restrict__ qo, __bf16* __restrict__ ko, __bf16* __restrict__ vo) {
    __shared__ __bf16 s_x[2][64 * 48];
    __shared__ __bf16 s_y[2][64 * 48];
    __shared__ __bf16 s_w[2][128 * 48];
    const int t = threadIdx.x, lane = t & 31, wave = t >> 5;
    const int r0 = blockIdx.x * 64;
    const int c0 = blockIdx.y * 128;
    const int wr = (wave >> 2) * 32, wc = (wave & 3) * 32;
    const int hi = (lane >> 4) & 1, lo = lane & 15;

#if USE_TDM
    if (wave == 0) {   // prologue: stage k0=0 tiles into buffer 0
        tdm_load_2d(lds_off(s_x[0]), xb + (long)r0 * CDIM, CDIM, 8192, CDIM, 32, 64, 3, 7);
        tdm_load_2d(lds_off(s_y[0]), y2b + (long)r0 * CDIM, CDIM, 8192, CDIM, 32, 64, 3, 7);
        tdm_load_2d(lds_off(s_w[0]), wb + (long)c0 * CDIM, CDIM, 3 * CDIM, CDIM, 32, 128, 3, 7);
    }
#endif

    v8f a1[2][2] = {}; v8f a2[2][2] = {};
    int cur = 0;
    for (int k0 = 0; k0 < CDIM; k0 += 32) {
        __syncthreads();
#if USE_TDM
        if (wave == 0) {
            if (k0 + 32 < CDIM) {
                int nb = cur ^ 1, kn = k0 + 32;
                tdm_load_2d(lds_off(s_x[nb]), xb + (long)r0 * CDIM + kn, CDIM, 8192, CDIM, 32, 64, 3, 7);
                tdm_load_2d(lds_off(s_y[nb]), y2b + (long)r0 * CDIM + kn, CDIM, 8192, CDIM, 32, 64, 3, 7);
                tdm_load_2d(lds_off(s_w[nb]), wb + (long)c0 * CDIM + kn, CDIM, 3 * CDIM, CDIM, 32, 128, 3, 7);
                __builtin_amdgcn_s_wait_tensorcnt(3);   // current tiles done; next 3 in flight
            } else {
                __builtin_amdgcn_s_wait_tensorcnt(0);
            }
        }
#else
        {
            int rr = t >> 2, seg = (t & 3) * 8;
            *(uint4*)&s_x[cur][rr * 48 + seg] = *(const uint4*)&xb [(long)(r0 + rr) * CDIM + k0 + seg];
            *(uint4*)&s_y[cur][rr * 48 + seg] = *(const uint4*)&y2b[(long)(r0 + rr) * CDIM + k0 + seg];
            *(uint4*)&s_w[cur][rr * 48 + seg]        = *(const uint4*)&wb[(long)(c0 + rr) * CDIM + k0 + seg];
            *(uint4*)&s_w[cur][(rr + 64) * 48 + seg] = *(const uint4*)&wb[(long)(c0 + 64 + rr) * CDIM + k0 + seg];
        }
#endif
        __syncthreads();
        const __bf16* px = s_x[cur];
        const __bf16* py = s_y[cur];
        const __bf16* pw = s_w[cur];
#pragma unroll
        for (int mi = 0; mi < 2; ++mi) {
            int row = wr + mi * 16 + lo;
            v16bf fa1 = make_frag(&px[row * 48 + hi * 8], &px[row * 48 + hi * 8 + 16]);
            v16bf fa2 = make_frag(&py[row * 48 + hi * 8], &py[row * 48 + hi * 8 + 16]);
#pragma unroll
            for (int ni = 0; ni < 2; ++ni) {
                int nr = wc + ni * 16 + lo;
                v16bf fb = make_frag(&pw[nr * 48 + hi * 8], &pw[nr * 48 + hi * 8 + 16]);
                a1[mi][ni] = __builtin_amdgcn_wmma_f32_16x16x32_bf16(
                    false, fa1, false, fb, (short)0, a1[mi][ni], false, false);
                a2[mi][ni] = __builtin_amdgcn_wmma_f32_16x16x32_bf16(
                    false, fa2, false, fb, (short)0, a2[mi][ni], false, false);
            }
        }
        cur ^= 1;
    }
    // epilogue: gate and scatter to q/k/v in [B,h,N,d] bf16
#pragma unroll
    for (int mi = 0; mi < 2; ++mi)
#pragma unroll
    for (int ni = 0; ni < 2; ++ni)
#pragma unroll
    for (int j = 0; j < 8; ++j) {
        int R = r0 + wr + mi * 16 + hi * 8 + j;
        int J = c0 + wc + ni * 16 + lo;
        int bb = R >> 10, n = R & (NTOK - 1);
        float bias = (J < CDIM) ? qbias[J] : kvbias[J - CDIM];
        float o1 = a1[mi][ni][j] + bias;
        float gt = sigmoidf_(a2[mi][ni][j] + bias);
        int c = (J < CDIM) ? J : (J < 2 * CDIM ? J - CDIM : J - 2 * CDIM);
        long idx = ((long)(bb * HEADS + (c >> 6)) * NTOK + n) * HD + (c & 63);
        if (J < CDIM)          qo[idx] = (__bf16)(o1 * (1.f + gt));
        else if (J < 2 * CDIM) ko[idx] = (__bf16)(gt * (o1 + 1.f));
        else                   vo[idx] = (__bf16)(gt * (o1 + 1.f));
    }
}

// ---------------------------------------------------------------- flash attention (per head)
__global__ __launch_bounds__(256)
void attn_kernel(const __bf16* __restrict__ qb, const __bf16* __restrict__ kb,
                 const __bf16* __restrict__ vb, __bf16* __restrict__ ob) {
    __shared__ __bf16 s_q[128 * 72];
    __shared__ __bf16 s_k[64 * 72];
    __shared__ __bf16 s_vt[64 * 72];   // V transposed: [d][key]
    __shared__ __bf16 s_p[8][16 * 72];
    const int t = threadIdx.x, lane = t & 31, wave = t >> 5;
    const int hi = (lane >> 4) & 1, lo = lane & 15;
    const int bh = blockIdx.x;
    const int q0 = blockIdx.y * 128;
    const __bf16* qh = qb + (long)bh * NTOK * HD;
    const __bf16* kh = kb + (long)bh * NTOK * HD;
    const __bf16* vh = vb + (long)bh * NTOK * HD;

#if USE_TDM
    if (wave == 0)     // stage Q tile 128x64 (LDS rows padded to 72)
        tdm_load_2d(lds_off(s_q), qh + (long)q0 * HD, HD, NTOK, HD, 64, 128, 4, 3);
#else
    {
        int seg = (t & 7) * 8, rb = t >> 3;
#pragma unroll
        for (int rep = 0; rep < 4; ++rep) {
            int r = rep * 32 + rb;
            *(uint4*)&s_q[r * 72 + seg] = *(const uint4*)&qh[(long)(q0 + r) * HD + seg];
        }
    }
#endif

    float m[8], l[8];
#pragma unroll
    for (int j = 0; j < 8; ++j) { m[j] = -1e30f; l[j] = 0.f; }
    v8f o[4] = {};
    const int qrow = wave * 16 + lo;
    const float sm = 0.125f * 1.44269504088896f;   // d^-0.5 * log2(e)

    for (int kt = 0; kt < NTOK; kt += 64) {
        __syncthreads();
#if USE_TDM
        if (wave == 0)
            tdm_load_2d(lds_off(s_k), kh + (long)kt * HD, HD, NTOK, HD, 64, 64, 4, 3);
#endif
        {   // V staged transposed through registers (TDM cannot transpose)
            int seg = (t & 7) * 8, rb = t >> 3;
#pragma unroll
            for (int rep = 0; rep < 2; ++rep) {
                int r = rep * 32 + rb;
#if !USE_TDM
                *(uint4*)&s_k[r * 72 + seg] = *(const uint4*)&kh[(long)(kt + r) * HD + seg];
#endif
                BF16x8 vv; vv.u = *(const uint4*)&vh[(long)(kt + r) * HD + seg];
#pragma unroll
                for (int i = 0; i < 8; ++i) s_vt[(seg + i) * 72 + r] = vv.v[i];
            }
        }
#if USE_TDM
        if (wave == 0) __builtin_amdgcn_s_wait_tensorcnt(0);
#endif
        __syncthreads();

        // S = Q K^T  (16 x 64 per wave)
        v8f s[4] = {};
#pragma unroll
        for (int ks = 0; ks < 2; ++ks) {
            v16bf fa = make_frag(&s_q[qrow * 72 + ks * 32 + hi * 8],
                                 &s_q[qrow * 72 + ks * 32 + hi * 8 + 16]);
#pragma unroll
            for (int ni = 0; ni < 4; ++ni) {
                int key = ni * 16 + lo;
                v16bf fb = make_frag(&s_k[key * 72 + ks * 32 + hi * 8],
                                     &s_k[key * 72 + ks * 32 + hi * 8 + 16]);
                s[ni] = __builtin_amdgcn_wmma_f32_16x16x32_bf16(
                    false, fa, false, fb, (short)0, s[ni], false, false);
            }
        }

        // online softmax: row r = hi*8+j lives in one 16-lane half -> shfl_xor reduce
#pragma unroll
        for (int j = 0; j < 8; ++j) {
            float mx = -1e30f;
#pragma unroll
            for (int ni = 0; ni < 4; ++ni) mx = fmaxf(mx, s[ni][j]);
#pragma unroll
            for (int off = 1; off < 16; off <<= 1) mx = fmaxf(mx, __shfl_xor(mx, off, 32));
            float mn = fmaxf(m[j], mx * sm);
            float corr = exp2f(m[j] - mn);
            float ps = 0.f;
#pragma unroll
            for (int ni = 0; ni < 4; ++ni) {
                float p = exp2f(s[ni][j] * sm - mn);
                s[ni][j] = p;
                ps += p;
            }
#pragma unroll
            for (int off = 1; off < 16; off <<= 1) ps += __shfl_xor(ps, off, 32);
            l[j] = l[j] * corr + ps;
            m[j] = mn;
#pragma unroll
            for (int ni = 0; ni < 4; ++ni) o[ni][j] *= corr;
        }

        // C-layout P -> wave-private LDS -> A-layout fragments
        __bf16* sp = &s_p[wave][0];
#pragma unroll
        for (int ni = 0; ni < 4; ++ni)
#pragma unroll
        for (int j = 0; j < 8; ++j)
            sp[(hi * 8 + j) * 72 + ni * 16 + lo] = (__bf16)s[ni][j];
        __syncthreads();

        // O += P @ V
#pragma unroll
        for (int ks = 0; ks < 2; ++ks) {
            v16bf fa = make_frag(&sp[lo * 72 + ks * 32 + hi * 8],
                                 &sp[lo * 72 + ks * 32 + hi * 8 + 16]);
#pragma unroll
            for (int ni = 0; ni < 4; ++ni) {
                int d = ni * 16 + lo;
                v16bf fb = make_frag(&s_vt[d * 72 + ks * 32 + hi * 8],
                                     &s_vt[d * 72 + ks * 32 + hi * 8 + 16]);
                o[ni] = __builtin_amdgcn_wmma_f32_16x16x32_bf16(
                    false, fa, false, fb, (short)0, o[ni], false, false);
            }
        }
    }

    // finalize: /l, stage to LDS, coalesced bf16 store in [B*N, C] layout
    __syncthreads();
    __bf16* sp = &s_p[wave][0];
#pragma unroll
    for (int ni = 0; ni < 4; ++ni)
#pragma unroll
    for (int j = 0; j < 8; ++j)
        sp[(hi * 8 + j) * 72 + ni * 16 + lo] = (__bf16)(o[ni][j] / l[j]);

    int bb = bh / HEADS, h = bh % HEADS;
    long base = ((long)bb * NTOK + q0 + wave * 16) * CDIM + h * HD;
    for (int e = lane; e < 16 * 64; e += 32) {
        int rr = e >> 6, cc = e & 63;
        ob[base + (long)rr * CDIM + cc] = sp[rr * 72 + cc];
    }
}

// ---------------------------------------------------------------- final projection GEMM (fp32 out + bias)
__global__ __launch_bounds__(256)
void gemm_proj_kernel(const __bf16* __restrict__ ab, const __bf16* __restrict__ wb,
                      const float* __restrict__ bias, float* __restrict__ out) {
    __shared__ __bf16 s_a[2][128 * 48];
    __shared__ __bf16 s_w[2][128 * 48];
    const int t = threadIdx.x, lane = t & 31, wave = t >> 5;
    const int r0 = blockIdx.x * 128, c0 = blockIdx.y * 128;
    const int wr = (wave >> 2) * 64, wc = (wave & 3) * 32;
    const int hi = (lane >> 4) & 1, lo = lane & 15;

#if USE_TDM
    if (wave == 0) {
        tdm_load_2d(lds_off(s_a[0]), ab + (long)r0 * CDIM, CDIM, 8192, CDIM, 32, 128, 3, 7);
        tdm_load_2d(lds_off(s_w[0]), wb + (long)c0 * CDIM, CDIM, CDIM, CDIM, 32, 128, 3, 7);
    }
#endif

    v8f acc[4][2] = {};
    int cur = 0;
    for (int k0 = 0; k0 < CDIM; k0 += 32) {
        __syncthreads();
#if USE_TDM
        if (wave == 0) {
            if (k0 + 32 < CDIM) {
                int nb = cur ^ 1, kn = k0 + 32;
                tdm_load_2d(lds_off(s_a[nb]), ab + (long)r0 * CDIM + kn, CDIM, 8192, CDIM, 32, 128, 3, 7);
                tdm_load_2d(lds_off(s_w[nb]), wb + (long)c0 * CDIM + kn, CDIM, CDIM, CDIM, 32, 128, 3, 7);
                __builtin_amdgcn_s_wait_tensorcnt(2);
            } else {
                __builtin_amdgcn_s_wait_tensorcnt(0);
            }
        }
#else
        {
            int rr = t >> 2, seg = (t & 3) * 8;
            *(uint4*)&s_a[cur][rr * 48 + seg]        = *(const uint4*)&ab[(long)(r0 + rr) * CDIM + k0 + seg];
            *(uint4*)&s_a[cur][(rr + 64) * 48 + seg] = *(const uint4*)&ab[(long)(r0 + 64 + rr) * CDIM + k0 + seg];
            *(uint4*)&s_w[cur][rr * 48 + seg]        = *(const uint4*)&wb[(long)(c0 + rr) * CDIM + k0 + seg];
            *(uint4*)&s_w[cur][(rr + 64) * 48 + seg] = *(const uint4*)&wb[(long)(c0 + 64 + rr) * CDIM + k0 + seg];
        }
#endif
        __syncthreads();
        const __bf16* pa = s_a[cur];
        const __bf16* pw = s_w[cur];
#pragma unroll
        for (int mi = 0; mi < 4; ++mi) {
            int row = wr + mi * 16 + lo;
            v16bf fa = make_frag(&pa[row * 48 + hi * 8], &pa[row * 48 + hi * 8 + 16]);
#pragma unroll
            for (int ni = 0; ni < 2; ++ni) {
                int nr = wc + ni * 16 + lo;
                v16bf fb = make_frag(&pw[nr * 48 + hi * 8], &pw[nr * 48 + hi * 8 + 16]);
                acc[mi][ni] = __builtin_amdgcn_wmma_f32_16x16x32_bf16(
                    false, fa, false, fb, (short)0, acc[mi][ni], false, false);
            }
        }
        cur ^= 1;
    }
#pragma unroll
    for (int mi = 0; mi < 4; ++mi)
#pragma unroll
    for (int ni = 0; ni < 2; ++ni)
#pragma unroll
    for (int j = 0; j < 8; ++j) {
        int R = r0 + wr + mi * 16 + hi * 8 + j;
        int J = c0 + wc + ni * 16 + lo;
        out[(long)R * CDIM + J] = acc[mi][ni][j] + bias[J];
    }
}

// ---------------------------------------------------------------- host launcher
extern "C" void kernel_launch(void* const* d_in, const int* in_sizes, int n_in,
                              void* d_out, int out_size, void* d_ws, size_t ws_size,
                              hipStream_t stream) {
    const float* x      = (const float*)d_in[0];
    const float* y      = (const float*)d_in[1];
    const float* q_w    = (const float*)d_in[2];
    const float* q_b    = (const float*)d_in[3];
    const float* kv_w   = (const float*)d_in[4];
    const float* kv_b   = (const float*)d_in[5];
    const float* proj_w = (const float*)d_in[6];
    const float* proj_b = (const float*)d_in[7];
    const float* ln_g   = (const float*)d_in[8];
    const float* ln_b   = (const float*)d_in[9];
    float* out = (float*)d_out;

    const long TOK = (long)BATCH * NTOK;            // 8192
    const long NX  = TOK * CDIM;                    // 6,291,456 elements
    char* ws = (char*)d_ws;
    size_t off = 0;
    __bf16* xb   = (__bf16*)(ws + off); off += NX * 2;               // reused as attn output
    __bf16* y2b  = (__bf16*)(ws + off); off += NX * 2;
    __bf16* wb   = (__bf16*)(ws + off); off += (size_t)3 * CDIM * CDIM * 2;
    __bf16* pwb  = (__bf16*)(ws + off); off += (size_t)CDIM * CDIM * 2;
    __bf16* qbuf = (__bf16*)(ws + off); off += NX * 2;
    __bf16* kbuf = (__bf16*)(ws + off); off += NX * 2;
    __bf16* vbuf = (__bf16*)(ws + off); off += NX * 2;
    __bf16* attn_o = xb;   // x is consumed by gemm_qkv before attention writes here

    cvt_bf16_kernel<<<2048, 256, 0, stream>>>(x, xb, (int)NX);
    cvt_bf16_kernel<<<512, 256, 0, stream>>>(q_w, wb, CDIM * CDIM);
    cvt_bf16_kernel<<<1024, 256, 0, stream>>>(kv_w, wb + (size_t)CDIM * CDIM, 2 * CDIM * CDIM);
    cvt_bf16_kernel<<<512, 256, 0, stream>>>(proj_w, pwb, CDIM * CDIM);

    ln_kernel<<<BATCH * (NTOK / 32), 256, 0, stream>>>(y, ln_g, ln_b, y2b);

    gemm_qkv_kernel<<<dim3(TOK / 64, (3 * CDIM) / 128), 256, 0, stream>>>(
        xb, y2b, wb, q_b, kv_b, qbuf, kbuf, vbuf);

    attn_kernel<<<dim3(BATCH * HEADS, NTOK / 128), 256, 0, stream>>>(qbuf, kbuf, vbuf, attn_o);

    gemm_proj_kernel<<<dim3(TOK / 128, CDIM / 128), 256, 0, stream>>>(attn_o, pwb, proj_b, out);
}
